// ptBEVnet_63204738728469
// MI455X (gfx1250) — matrix operand order repaired and checked by reading, not compiled, not run on hardware
//
#include <hip/hip_runtime.h>
#include <math.h>

// ---------------- problem constants ----------------
#define EPSV 1e-5f
constexpr int Bsz  = 4;
constexpr int Npts = 100000;
constexpr int FD   = 9;
constexpr int Hg   = 480;
constexpr int Wg   = 360;
constexpr int COMPD = 32;
constexpr int NSEG = Bsz * Hg * Wg;          // 691200
constexpr int NPTS_TOT = Bsz * Npts;         // 400000

typedef __attribute__((ext_vector_type(16))) _Float16 v16h;
typedef __attribute__((ext_vector_type(8)))  _Float16 v8h;
typedef __attribute__((ext_vector_type(8)))  float    v8f;

// order-preserving float <-> uint encoding for atomic max on floats
__device__ __forceinline__ unsigned encf(float f) {
    unsigned u = __float_as_uint(f);
    return (u & 0x80000000u) ? ~u : (u | 0x80000000u);
}
__device__ __forceinline__ float decf(unsigned u) {
    unsigned b = (u & 0x80000000u) ? (u & 0x7FFFFFFFu) : ~u;
    return __uint_as_float(b);
}

// ---------------- weight prep: BN-fold + swizzle into WMMA B-fragment order ----
// dst element t maps to (nt,kt,lane,e); source (k,n) with
//   k = kt*32 + 8*(lane>=16) + (e&7) + 16*(e>=8)   (16-bit B 32x16 lane layout)
//   n = nt*16 + (lane&15)
__global__ void swizzle_w(const float* __restrict__ W, _Float16* __restrict__ dst,
                          int Kd, int Nd, int Ksrc,
                          const float* gR, const float* vR,   // optional row scale (BN folded into input)
                          const float* gC, const float* vC) { // optional col scale (BN folded into output)
    int t = blockIdx.x * blockDim.x + threadIdx.x;
    int tot = Kd * Nd;
    if (t >= tot) return;
    int e    = t & 15;
    int lane = (t >> 4) & 31;
    int tile = t >> 9;
    int ktiles = Kd >> 5;
    int nt = tile / ktiles;
    int kt = tile - nt * ktiles;
    int k = kt * 32 + ((lane >= 16) ? 8 : 0) + (e & 7) + ((e >= 8) ? 16 : 0);
    int n = nt * 16 + (lane & 15);
    float val = 0.f;
    if (k < Ksrc) {
        val = W[k * Nd + n];
        if (gR) val *= gR[k] * rsqrtf(vR[k] + EPSV);
        if (gC) val *= gC[n] * rsqrtf(vC[n] + EPSV);
    }
    dst[t] = (_Float16)val;
}

// folded biases: [0,64) b1', [64,192) b2', [192,448) b3', [448,704) b4, [704,736) bc
__global__ void fold_bias(const float* g0, const float* be0, const float* m0, const float* vv0,
                          const float* g1, const float* be1, const float* m1, const float* vv1,
                          const float* g2, const float* be2, const float* m2, const float* vv2,
                          const float* g3, const float* be3, const float* m3, const float* vv3,
                          const float* W1, const float* b1, const float* b2,
                          const float* b3, const float* b4, const float* bc,
                          float* __restrict__ out) {
    int j = threadIdx.x;
    if (j < 64) {
        float s1 = g1[j] * rsqrtf(vv1[j] + EPSV);
        float t1 = be1[j] - m1[j] * s1;
        float acc = b1[j];
        for (int i = 0; i < FD; ++i) {
            float s0 = g0[i] * rsqrtf(vv0[i] + EPSV);
            float t0 = be0[i] - m0[i] * s0;
            acc += t0 * W1[i * 64 + j];
        }
        out[j] = acc * s1 + t1;
    }
    if (j < 128) {
        float s2 = g2[j] * rsqrtf(vv2[j] + EPSV);
        out[64 + j] = b2[j] * s2 + (be2[j] - m2[j] * s2);
    }
    if (j < 256) {
        float s3 = g3[j] * rsqrtf(vv3[j] + EPSV);
        out[192 + j] = b3[j] * s3 + (be3[j] - m3[j] * s3);
        out[448 + j] = b4[j];
    }
    if (j < 32) out[704 + j] = bc[j];
}

// ---------------- per-wave fused GEMM step (LDS -> LDS), M = MT*16 rows -------
// 16-bit A 16x32 lane layout: lane<16 holds K 0..7,16..23 ; lane>=16 holds K 8..15,24..31
template<int Kd, int Nd, int MT, bool RELU>
__device__ __forceinline__ void wave_gemm_lds(const _Float16* actIn, _Float16* actOut,
                                              const _Float16* __restrict__ Wswz,
                                              const float* __restrict__ bias, int lane) {
    const int col   = lane & 15;
    const int row   = lane & 15;
    const int rbase = (lane >= 16) ? 8 : 0;
    const int koff  = (lane >= 16) ? 8 : 0;
#pragma unroll
    for (int nt = 0; nt < Nd / 16; ++nt) {
        v8f c[MT] = {};
#pragma unroll
        for (int kt = 0; kt < Kd / 32; ++kt) {
            v16h b = *(const v16h*)(Wswz + ((size_t)(nt * (Kd / 32) + kt) * 32 + lane) * 16);
#pragma unroll
            for (int mt = 0; mt < MT; ++mt) {
                const _Float16* ap = actIn + (mt * 16 + row) * Kd + kt * 32 + koff;
                v8h lo = *(const v8h*)(ap);
                v8h hi = *(const v8h*)(ap + 16);
                v16h a;
#pragma unroll
                for (int i = 0; i < 8; ++i) { a[i] = lo[i]; a[i + 8] = hi[i]; }
                c[mt] = __builtin_amdgcn_wmma_f32_16x16x32_f16(false, a, false, b,
                                                               (short)0, c[mt], false, false);
            }
        }
        float bb = bias[nt * 16 + col];
#pragma unroll
        for (int mt = 0; mt < MT; ++mt) {
            _Float16* outp = actOut + (mt * 16 + rbase) * Nd + nt * 16 + col;
#pragma unroll
            for (int v = 0; v < 8; ++v) {
                float val = c[mt][v] + bb;
                if (RELU) val = fmaxf(val, 0.f);
                outp[v * Nd] = (_Float16)val;   // constant DS immediate offsets
            }
        }
    }
}

// ---------------- fused point MLP + scatter-max (32 points per wave) ----------
__global__ void __launch_bounds__(64) mlp_kernel(
    const float* __restrict__ pt_fea, const int* __restrict__ xy_ind,
    const _Float16* __restrict__ W1s, const _Float16* __restrict__ W2s,
    const _Float16* __restrict__ W3s, const _Float16* __restrict__ W4s,
    const float* __restrict__ biases,
    unsigned* __restrict__ pooled, unsigned* __restrict__ cnt)
{
    __shared__ _Float16 sbuf0[2][32 * 128];   // 32x32 (L1 in) / 32x128 (L2 out)
    __shared__ _Float16 sbuf1[2][32 * 256];   // 32x64  (L1 out) / 32x256 (L3 out)
    __shared__ int sfid[2][32];
    const int lane = threadIdx.x & 31;
    const int wv   = threadIdx.x >> 5;
    const int tile = blockIdx.x * 2 + wv;
    const int base = tile * 32;
    _Float16* buf0 = sbuf0[wv];
    _Float16* buf1 = sbuf1[wv];

    // warm caches for the shared weight arrays (global_prefetch_b8)
    __builtin_prefetch(W2s + (size_t)lane * 256, 0, 1);
    __builtin_prefetch(W3s + (size_t)lane * 1024, 0, 1);
    __builtin_prefetch(W4s + (size_t)lane * 2048, 0, 1);

    // stage 32x9 input into 32x32 (zero-padded). BN0 is folded into W1s.
    {
        int p = base + lane;
        const float* src = pt_fea + (size_t)p * FD;
#pragma unroll
        for (int k = 0; k < FD; ++k) buf0[lane * 32 + k] = (_Float16)src[k];
#pragma unroll
        for (int k = FD; k < 32; ++k) buf0[lane * 32 + k] = (_Float16)0.f;
        int b  = p / Npts;
        int gx = xy_ind[2 * p];
        int gy = xy_ind[2 * p + 1];
        sfid[wv][lane] = (b * Hg + gx) * Wg + gy;
    }
    // same-wave LDS ops are in order (DScnt in-order per wave) -> no barrier needed.

    wave_gemm_lds<32,  64,  2, true>(buf0, buf1, W1s, biases + 0,   lane);
    wave_gemm_lds<64,  128, 2, true>(buf1, buf0, W2s, biases + 64,  lane);
    wave_gemm_lds<128, 256, 2, true>(buf0, buf1, W3s, biases + 192, lane);

    // final 256->256 layer (no relu): scatter directly via encoded atomic max
    {
        const float* b4 = biases + 448;
        const int col   = lane & 15;
        const int row   = lane & 15;
        const int rbase = (lane >= 16) ? 8 : 0;
        const int koff  = (lane >= 16) ? 8 : 0;
#pragma unroll
        for (int nt = 0; nt < 16; ++nt) {
            v8f c[2] = {};
#pragma unroll
            for (int kt = 0; kt < 8; ++kt) {
                v16h b = *(const v16h*)(W4s + ((size_t)(nt * 8 + kt) * 32 + lane) * 16);
#pragma unroll
                for (int mt = 0; mt < 2; ++mt) {
                    const _Float16* ap = buf1 + (mt * 16 + row) * 256 + kt * 32 + koff;
                    v8h lo = *(const v8h*)(ap);
                    v8h hi = *(const v8h*)(ap + 16);
                    v16h a;
#pragma unroll
                    for (int i = 0; i < 8; ++i) { a[i] = lo[i]; a[i + 8] = hi[i]; }
                    c[mt] = __builtin_amdgcn_wmma_f32_16x16x32_f16(false, a, false, b,
                                                                   (short)0, c[mt], false, false);
                }
            }
            float bb = b4[nt * 16 + col];
#pragma unroll
            for (int mt = 0; mt < 2; ++mt) {
#pragma unroll
                for (int v = 0; v < 8; ++v) {
                    int r  = mt * 16 + rbase + v;
                    int fi = sfid[wv][r];
                    unsigned e = encf(c[mt][v] + bb);
                    atomicMax(&pooled[(size_t)fi * 256 + nt * 16 + col], e);
                }
            }
        }
    }
    atomicAdd(&cnt[sfid[wv][lane]], 1u);
}

// ---------------- compression GEMM (256 -> 32) on voxels, writes [B,C,H,W] ----
__global__ void __launch_bounds__(64) compress_kernel(
    const unsigned* __restrict__ pooled, const unsigned* __restrict__ cnt,
    const _Float16* __restrict__ Wcs, const float* __restrict__ biases,
    float* __restrict__ comp)
{
    __shared__ _Float16 sact[2][32 * 256];
    const int lane  = threadIdx.x & 31;
    const int wv    = threadIdx.x >> 5;
    const int tile  = blockIdx.x * 2 + wv;
    const int sbase = tile * 32;
    _Float16* act = sact[wv];

    const unsigned* src = pooled + (size_t)sbase * 256;
#pragma unroll 8
    for (int i = lane; i < 32 * 256; i += 32)
        act[i] = (_Float16)decf(src[i]);   // empty voxels decode to NaN; gated by occ below

    const float* bc = biases + 704;
    const int col   = lane & 15;
    const int row   = lane & 15;
    const int rbase = (lane >= 16) ? 8 : 0;
    const int koff  = (lane >= 16) ? 8 : 0;
#pragma unroll
    for (int nt = 0; nt < 2; ++nt) {
        v8f c[2] = {};
#pragma unroll
        for (int kt = 0; kt < 8; ++kt) {
            v16h b = *(const v16h*)(Wcs + ((size_t)(nt * 8 + kt) * 32 + lane) * 16);
#pragma unroll
            for (int mt = 0; mt < 2; ++mt) {
                const _Float16* ap = act + (mt * 16 + row) * 256 + kt * 32 + koff;
                v8h lo = *(const v8h*)(ap);
                v8h hi = *(const v8h*)(ap + 16);
                v16h a;
#pragma unroll
                for (int i = 0; i < 8; ++i) { a[i] = lo[i]; a[i + 8] = hi[i]; }
                c[mt] = __builtin_amdgcn_wmma_f32_16x16x32_f16(false, a, false, b,
                                                               (short)0, c[mt], false, false);
            }
        }
        float bb = bc[nt * 16 + col];
        int ch = nt * 16 + col;
#pragma unroll
        for (int mt = 0; mt < 2; ++mt) {
#pragma unroll
            for (int v = 0; v < 8; ++v) {
                int seg = sbase + mt * 16 + rbase + v;
                bool occ = cnt[seg] > 0;
                float val = occ ? fmaxf(c[mt][v] + bb, 0.f) : 0.f;
                int b  = seg / (Hg * Wg);
                int hw = seg - b * (Hg * Wg);
                comp[((size_t)(b * COMPD + ch)) * (Hg * Wg) + hw] = val;
            }
        }
    }
}

// ---------------- 3x3 stride-1 pad-1 max pool ----------------
__global__ void maxpool_kernel(const float* __restrict__ comp, float* __restrict__ out, int total)
{
    int i = blockIdx.x * blockDim.x + threadIdx.x;
    if (i >= total) return;
    int w = i % Wg; int t = i / Wg;
    int h = t % Hg; t /= Hg;                 // t = b*COMPD + ch
    const float* plane = comp + (size_t)t * (Hg * Wg);
    float m = -3.402823466e+38f;
#pragma unroll
    for (int dh = -1; dh <= 1; ++dh) {
        int hh = h + dh; if (hh < 0 || hh >= Hg) continue;
#pragma unroll
        for (int dw = -1; dw <= 1; ++dw) {
            int ww = w + dw; if (ww < 0 || ww >= Wg) continue;
            m = fmaxf(m, plane[hh * Wg + ww]);
        }
    }
    out[i] = m;
}

extern "C" void kernel_launch(void* const* d_in, const int* in_sizes, int n_in,
                              void* d_out, int out_size, void* d_ws, size_t ws_size,
                              hipStream_t stream)
{
    (void)in_sizes; (void)n_in; (void)out_size; (void)ws_size;
    const float* pt_fea = (const float*)d_in[0];
    const int*   xy_ind = (const int*)d_in[1];
    const float *bn0g = (const float*)d_in[2],  *bn0b = (const float*)d_in[3],
                *bn0m = (const float*)d_in[4],  *bn0v = (const float*)d_in[5];
    const float *bn1g = (const float*)d_in[6],  *bn1b = (const float*)d_in[7],
                *bn1m = (const float*)d_in[8],  *bn1v = (const float*)d_in[9];
    const float *bn2g = (const float*)d_in[10], *bn2b = (const float*)d_in[11],
                *bn2m = (const float*)d_in[12], *bn2v = (const float*)d_in[13];
    const float *bn3g = (const float*)d_in[14], *bn3b = (const float*)d_in[15],
                *bn3m = (const float*)d_in[16], *bn3v = (const float*)d_in[17];
    const float *W1 = (const float*)d_in[18], *b1 = (const float*)d_in[19];
    const float *W2 = (const float*)d_in[20], *b2 = (const float*)d_in[21];
    const float *W3 = (const float*)d_in[22], *b3 = (const float*)d_in[23];
    const float *W4 = (const float*)d_in[24], *b4 = (const float*)d_in[25];
    const float *Wc = (const float*)d_in[26], *bc = (const float*)d_in[27];

    char* ws = (char*)d_ws;
    size_t off = 0;
    _Float16* W1s = (_Float16*)(ws + off); off += (size_t)32 * 64 * 2;
    _Float16* W2s = (_Float16*)(ws + off); off += (size_t)64 * 128 * 2;
    _Float16* W3s = (_Float16*)(ws + off); off += (size_t)128 * 256 * 2;
    _Float16* W4s = (_Float16*)(ws + off); off += (size_t)256 * 256 * 2;
    _Float16* Wcs = (_Float16*)(ws + off); off += (size_t)256 * 32 * 2;
    float* biases = (float*)(ws + off);    off += (size_t)736 * 4;
    off = (off + 255) & ~(size_t)255;
    unsigned* cnt = (unsigned*)(ws + off);    off += (size_t)NSEG * 4;
    off = (off + 255) & ~(size_t)255;
    unsigned* pooled = (unsigned*)(ws + off); off += (size_t)NSEG * 256 * 4;
    off = (off + 255) & ~(size_t)255;
    float* comp = (float*)(ws + off);         off += (size_t)NSEG * COMPD * 4;

    // reset reduction buffers each call (encoded 0 == "-inf" sentinel)
    hipMemsetAsync(pooled, 0, (size_t)NSEG * 256 * 4, stream);
    hipMemsetAsync(cnt,    0, (size_t)NSEG * 4,       stream);

    // weight prep (tiny): BN fold + fragment swizzle to f16
    swizzle_w<<<(32 * 64   + 255) / 256, 256, 0, stream>>>(W1, W1s, 32, 64,  FD,  bn0g, bn0v, bn1g, bn1v);
    swizzle_w<<<(64 * 128  + 255) / 256, 256, 0, stream>>>(W2, W2s, 64, 128, 64,  nullptr, nullptr, bn2g, bn2v);
    swizzle_w<<<(128 * 256 + 255) / 256, 256, 0, stream>>>(W3, W3s, 128, 256, 128, nullptr, nullptr, bn3g, bn3v);
    swizzle_w<<<(256 * 256 + 255) / 256, 256, 0, stream>>>(W4, W4s, 256, 256, 256, nullptr, nullptr, nullptr, nullptr);
    swizzle_w<<<(256 * 32  + 255) / 256, 256, 0, stream>>>(Wc, Wcs, 256, 32,  256, nullptr, nullptr, nullptr, nullptr);
    fold_bias<<<1, 256, 0, stream>>>(bn0g, bn0b, bn0m, bn0v, bn1g, bn1b, bn1m, bn1v,
                                     bn2g, bn2b, bn2m, bn2v, bn3g, bn3b, bn3m, bn3v,
                                     W1, b1, b2, b3, b4, bc, biases);

    // fused MLP + scatter-max: 2 waves/block, one 32-point tile per wave
    mlp_kernel<<<NPTS_TOT / 64, 64, 0, stream>>>(pt_fea, xy_ind, W1s, W2s, W3s, W4s,
                                                 biases, pooled, cnt);
    // 256->32 compression with WMMA (32 segments per wave), writes [B,C,H,W]
    compress_kernel<<<NSEG / 64, 64, 0, stream>>>(pooled, cnt, Wcs, biases, comp);
    // 3x3 max pool
    int total = Bsz * COMPD * Hg * Wg;
    maxpool_kernel<<<(total + 255) / 256, 256, 0, stream>>>(comp, (float*)d_out, total);
}
